// GL_TAGConv_9l_128h_w_k3_44753559224344
// MI455X (gfx1250) — compile-verified, matrix operand order checked by reading, not run on hardware
//
#include <hip/hip_runtime.h>
#include <hip/hip_bf16.h>

typedef __attribute__((ext_vector_type(16))) _Float16 v16h;
typedef __attribute__((ext_vector_type(8)))  _Float16 v8h;
typedef __attribute__((ext_vector_type(4)))  _Float16 v4h;
typedef __attribute__((ext_vector_type(8)))  float    v8f;
typedef __attribute__((ext_vector_type(4)))  unsigned int v4u;
typedef __attribute__((ext_vector_type(8)))  int      v8i;
typedef __attribute__((ext_vector_type(4)))  int      v4i;

#define FDIM 128
#define KHOP 3
#define KCAT 512   // (KHOP+1)*FDIM
#define NLAYERS 8
#define COUT 10
#define SCAN_T 1024

static inline int cdiv(long long a, long long b) { return (int)((a + b - 1) / b); }

// ---------------- small utility kernels ----------------

__global__ void zero_i32(int* __restrict__ p, int n) {
  int i = blockIdx.x * blockDim.x + threadIdx.x;
  if (i < n) p[i] = 0;
}

__global__ void zero_f32(float* __restrict__ p, int n) {
  int i = blockIdx.x * blockDim.x + threadIdx.x;
  if (i < n) p[i] = 0.0f;
}

__global__ void deg_kernel(const int* __restrict__ dst, const float* __restrict__ w,
                           float* __restrict__ deg, int E) {
  int e = blockIdx.x * blockDim.x + threadIdx.x;
  if (e < E) atomicAdd(&deg[dst[e]], w[e]);
}

__global__ void dis_kernel(float* __restrict__ deg, int n) {
  int i = blockIdx.x * blockDim.x + threadIdx.x;
  if (i < n) {
    float d = deg[i];
    deg[i] = (d > 0.0f) ? rsqrtf(d) : 0.0f;
  }
}

__global__ void norm_kernel(const int* __restrict__ src, const int* __restrict__ dst,
                            const float* __restrict__ w, const float* __restrict__ dis,
                            float* __restrict__ norm, int E) {
  int e = blockIdx.x * blockDim.x + threadIdx.x;
  if (e < E) norm[e] = dis[src[e]] * w[e] * dis[dst[e]];
}

// ---------------- CSR construction (by destination) ----------------

__global__ void count_kernel(const int* __restrict__ dst, int* __restrict__ cnt, int E) {
  int e = blockIdx.x * blockDim.x + threadIdx.x;
  if (e < E) atomicAdd(&cnt[dst[e]], 1);
}

__global__ __launch_bounds__(SCAN_T) void scan_kernel(const int* __restrict__ cnt,
                                                      int* __restrict__ rowptr,
                                                      int* __restrict__ fill, int N) {
  __shared__ int part[SCAN_T];
  const int tid = threadIdx.x;
  const int chunk = (N + SCAN_T - 1) / SCAN_T;
  int beg = tid * chunk;
  int end = beg + chunk; if (end > N) end = N;
  if (beg > N) beg = N;

  int s = 0;
  for (int i = beg; i < end; ++i) s += cnt[i];
  part[tid] = s;
  __syncthreads();
  for (int off = 1; off < SCAN_T; off <<= 1) {
    int v = part[tid];
    int add = (tid >= off) ? part[tid - off] : 0;
    __syncthreads();
    part[tid] = v + add;
    __syncthreads();
  }
  int run = (tid == 0) ? 0 : part[tid - 1];
  for (int i = beg; i < end; ++i) {
    rowptr[i] = run;
    fill[i]   = run;
    run += cnt[i];
  }
  if (tid == SCAN_T - 1) rowptr[N] = part[SCAN_T - 1];
}

__global__ void scatter_kernel(const int* __restrict__ src, const int* __restrict__ dst,
                               const float* __restrict__ norm, int* __restrict__ fill,
                               int* __restrict__ colsrc, float* __restrict__ colw, int E) {
  int e = blockIdx.x * blockDim.x + threadIdx.x;
  if (e >= E) return;
  int p = atomicAdd(&fill[dst[e]], 1);
  colsrc[p] = src[e];
  colw[p]   = norm[e];
}

// ---------------- weight transpose/convert ----------------

__global__ void wcvt_hidden(const float* __restrict__ W, _Float16* __restrict__ Wt, int total) {
  int idx = blockIdx.x * blockDim.x + threadIdx.x;   // total = 8*512*128
  if (idx >= total) return;
  int l = idx >> 16;
  int r = idx & 65535;
  int k = r >> 7;
  int n = r & 127;
  Wt[((size_t)l << 16) + (size_t)n * KCAT + k] =
      (_Float16)W[((size_t)l << 16) + (size_t)k * FDIM + n];
}

__global__ void wcvt_out(const float* __restrict__ W, _Float16* __restrict__ Wt, int total) {
  int idx = blockIdx.x * blockDim.x + threadIdx.x;   // total = 16*512
  if (idx >= total) return;
  int n = idx >> 9;
  int k = idx & 511;
  Wt[idx] = (n < COUT) ? (_Float16)W[(size_t)k * COUT + n] : (_Float16)0.0f;
}

__global__ void cast_slice0(const float* __restrict__ h, _Float16* __restrict__ Hcat, int total) {
  int idx = blockIdx.x * blockDim.x + threadIdx.x;   // total = N*128
  if (idx >= total) return;
  int nrow = idx >> 7;
  int f = idx & 127;
  Hcat[(size_t)nrow * KCAT + f] = (_Float16)h[idx];
}

// ---------------- gather SpMM (no atomics) + fused f16 slice write ----------------

__global__ void spmm_csr(const int* __restrict__ rowptr, const int* __restrict__ colsrc,
                         const float* __restrict__ colw, const float* __restrict__ hin,
                         float* __restrict__ hout, _Float16* __restrict__ hslice, int N) {
  int w = (blockIdx.x * blockDim.x + threadIdx.x) >> 5;
  int lane = threadIdx.x & 31;
  if (w >= N) return;
  int beg = rowptr[w];
  int end = rowptr[w + 1];
  float ax = 0.f, ay = 0.f, az = 0.f, aw = 0.f;
  for (int i = beg; i < end; ++i) {
    int   s = colsrc[i];   // uniform across wave -> scalarized broadcast
    float c = colw[i];
    float4 v = ((const float4*)(hin + (size_t)s * FDIM))[lane];
    ax += c * v.x; ay += c * v.y; az += c * v.z; aw += c * v.w;
  }
  float4 r; r.x = ax; r.y = ay; r.z = az; r.w = aw;
  ((float4*)(hout + (size_t)w * FDIM))[lane] = r;
  v4h h; h[0] = (_Float16)ax; h[1] = (_Float16)ay; h[2] = (_Float16)az; h[3] = (_Float16)aw;
  *(v4h*)(hslice + (size_t)w * KCAT + lane * 4) = h;
}

// ---------------- TDM helper: stage weight matrix into LDS ----------------
// 1-D tile of 8-byte elements, per ISA D# groups (CDNA5 ISA).
// This toolchain exposes the 6-arg builtin: (v4u g0, v8i g1, v4i g2, v4i g3, v8i, i32 cpol)
__device__ __forceinline__ void tdm_stage_to_lds(const _Float16* gsrc, unsigned lds_off,
                                                 unsigned nbytes) {
  unsigned long long ga = (unsigned long long)(uintptr_t)gsrc;
  unsigned n8 = nbytes >> 3;              // 8-byte elements
  v4u g0;
  g0[0] = 1u;                                          // count=1, user mode
  g0[1] = lds_off;                                     // lds_addr (bytes)
  g0[2] = (unsigned)(ga & 0xFFFFFFFFu);                // global_addr[31:0]
  g0[3] = (unsigned)((ga >> 32) & 0x1FFFFFFu) | (2u << 30);  // addr[56:32] | type=2
  v8i g1;
  g1[0] = 0x00030000;                                  // data_size=3 (8B), mask=0
  g1[1] = (int)((n8 & 0xFFFFu) << 16);                 // tensor_dim0[15:0]
  g1[2] = (int)(((n8 >> 16) & 0xFFFFu) | (1u << 16));  // tensor_dim0[31:16] | tensor_dim1=1
  g1[3] = (int)((n8 & 0xFFFFu) << 16);                 // tile_dim0 = n8 (fits 16 bits)
  g1[4] = 0;                                           // tile_dim1=0 (unused), tile_dim2=0
  g1[5] = (int)n8;                                     // tensor_dim0_stride[31:0]
  g1[6] = 0;
  g1[7] = 0;
  v4i gz4 = {0, 0, 0, 0};
  v8i gz8 = {0, 0, 0, 0, 0, 0, 0, 0};
  __builtin_amdgcn_tensor_load_to_lds(g0, g1, gz4, gz4, gz8, 0);
}

// ---------------- WMMA GEMM with TDM-staged LDS weights ----------------
// Block = 8 waves; wave owns one 16-row tile, iterates NCT column tiles.
// B fragments come from LDS (ds_load), A fragments from global (vmem pipe).
template <int NCT>
__global__ __launch_bounds__(256) void gemm_tag(const _Float16* __restrict__ A,
                                                const _Float16* __restrict__ Bt,
                                                const float* __restrict__ bias,
                                                float* __restrict__ out,
                                                _Float16* __restrict__ hnext,
                                                int nrowtiles, int out_stride,
                                                int ncols_valid, int do_elu) {
  extern __shared__ _Float16 Bs[];        // NCT*16*KCAT halves

  // --- stage weights into LDS via Tensor Data Mover (wave 0 issues; EXEC ignored) ---
  if (threadIdx.x < 32) {
    unsigned lds_base = (unsigned)__builtin_amdgcn_groupstaticsize();
    tdm_stage_to_lds(Bt, lds_base, (unsigned)(NCT * 16 * KCAT * sizeof(_Float16)));
    __builtin_amdgcn_s_wait_tensorcnt(0);
  }
  __syncthreads();

  const int wave = threadIdx.x >> 5;
  const int lane = threadIdx.x & 31;
  const int half = lane >> 4;
  const int nn   = lane & 15;
  const int rowtile = blockIdx.x * 8 + wave;
  if (rowtile >= nrowtiles) return;       // uniform per wave: EXEC stays all-ones
  const int rowbase = rowtile * 16;

  const _Float16* arow = A + (size_t)(rowbase + nn) * KCAT;

  v8f acc[NCT];
#pragma unroll
  for (int ct = 0; ct < NCT; ++ct) acc[ct] = (v8f){};

#pragma unroll
  for (int kb = 0; kb < KCAT; kb += 32) {
    const int k0 = kb + half * 8;
    // 16-bit A-matrix VGPR layout: regs 0-3 hold K = half*8+{0..7}, regs 4-7 hold +16
    v8h a_lo = *(const v8h*)(arow + k0);
    v8h a_hi = *(const v8h*)(arow + k0 + 16);
    v16h a = __builtin_shufflevector(a_lo, a_hi, 0,1,2,3,4,5,6,7,8,9,10,11,12,13,14,15);
#pragma unroll
    for (int ct = 0; ct < NCT; ++ct) {
      const int boff = (ct * 16 + nn) * KCAT + k0;
      v8h b_lo = *(const v8h*)&Bs[boff];
      v8h b_hi = *(const v8h*)&Bs[boff + 16];
      v16h b = __builtin_shufflevector(b_lo, b_hi, 0,1,2,3,4,5,6,7,8,9,10,11,12,13,14,15);
      acc[ct] = __builtin_amdgcn_wmma_f32_16x16x32_f16(false, a, false, b, (short)0,
                                                       acc[ct], false, false);
    }
  }

#pragma unroll
  for (int ct = 0; ct < NCT; ++ct) {
    const int col = ct * 16 + nn;
    const bool colok = (col < ncols_valid);
    const float bv = colok ? bias[col] : 0.0f;
#pragma unroll
    for (int r = 0; r < 8; ++r) {
      int m = rowbase + r + 8 * half;
      float v = acc[ct][r] + bv;
      if (do_elu) v = (v > 0.0f) ? v : (__expf(v) - 1.0f);
      if (colok) {
        out[(size_t)m * out_stride + col] = v;
        if (hnext) hnext[(size_t)m * KCAT + col] = (_Float16)v;
      }
    }
  }
}

// ---------------- host orchestration ----------------

extern "C" void kernel_launch(void* const* d_in, const int* in_sizes, int n_in,
                              void* d_out, int out_size, void* d_ws, size_t ws_size,
                              hipStream_t stream) {
  const float* x         = (const float*)d_in[0];
  const int*   ei        = (const int*)  d_in[1];
  const float* edge_attr = (const float*)d_in[2];
  const float* W_hidden  = (const float*)d_in[3];
  const float* b_hidden  = (const float*)d_in[4];
  const float* W_out     = (const float*)d_in[5];
  const float* b_out     = (const float*)d_in[6];

  const int N = in_sizes[0] / FDIM;
  const int E = in_sizes[2];
  const int* src = ei;
  const int* dst = ei + E;

  char* ws = (char*)d_ws;
  size_t off = 0;
  auto carve = [&](size_t bytes) -> void* {
    void* p = ws + off;
    off = (off + bytes + 255) & ~(size_t)255;
    return p;
  };
  float*    norm   = (float*)   carve((size_t)E * 4);
  float*    deg    = (float*)   carve((size_t)N * 4);
  int*      cnt    = (int*)     carve((size_t)N * 4);
  int*      rowptr = (int*)     carve((size_t)(N + 1) * 4);
  int*      fill   = (int*)     carve((size_t)N * 4);
  int*      colsrc = (int*)     carve((size_t)E * 4);
  float*    colw   = (float*)   carve((size_t)E * 4);
  float*    h0     = (float*)   carve((size_t)N * FDIM * 4);
  float*    h1     = (float*)   carve((size_t)N * FDIM * 4);
  _Float16* HcatA  = (_Float16*)carve((size_t)N * KCAT * 2);
  _Float16* HcatB  = (_Float16*)carve((size_t)N * KCAT * 2);
  _Float16* Wt16   = (_Float16*)carve((size_t)NLAYERS * FDIM * KCAT * 2);
  _Float16* WoutT  = (_Float16*)carve((size_t)16 * KCAT * 2);
  (void)ws_size;

  const int NF = N * FDIM;
  const int nrowtiles = N / 16;                 // 6250 (exact)
  const int gemm_blocks = cdiv(nrowtiles, 8);   // 782
  const size_t lds_hidden = (size_t)128 * KCAT * sizeof(_Float16); // 128 KB
  const size_t lds_out    = (size_t)16  * KCAT * sizeof(_Float16); // 16 KB

  // allow 128 KB dynamic LDS (host-side attribute, capture-safe)
  (void)hipFuncSetAttribute((const void*)&gemm_tag<8>,
                            hipFuncAttributeMaxDynamicSharedMemorySize, (int)lds_hidden);
  (void)hipFuncSetAttribute((const void*)&gemm_tag<1>,
                            hipFuncAttributeMaxDynamicSharedMemorySize, (int)lds_out);

  // 1) gcn_norm
  zero_f32<<<cdiv(N, 256), 256, 0, stream>>>(deg, N);
  deg_kernel<<<cdiv(E, 256), 256, 0, stream>>>(dst, edge_attr, deg, E);
  dis_kernel<<<cdiv(N, 256), 256, 0, stream>>>(deg, N);
  norm_kernel<<<cdiv(E, 256), 256, 0, stream>>>(src, dst, edge_attr, deg, norm, E);

  // 2) CSR by destination
  zero_i32<<<cdiv(N, 256), 256, 0, stream>>>(cnt, N);
  count_kernel<<<cdiv(E, 256), 256, 0, stream>>>(dst, cnt, E);
  scan_kernel<<<1, SCAN_T, 0, stream>>>(cnt, rowptr, fill, N);
  scatter_kernel<<<cdiv(E, 256), 256, 0, stream>>>(src, dst, norm, fill, colsrc, colw, E);

  // 3) weights -> transposed f16
  wcvt_hidden<<<cdiv(NLAYERS * KCAT * FDIM, 256), 256, 0, stream>>>(W_hidden, Wt16,
                                                                    NLAYERS * KCAT * FDIM);
  wcvt_out<<<cdiv(16 * KCAT, 256), 256, 0, stream>>>(W_out, WoutT, 16 * KCAT);

  // 4) initial slice 0
  cast_slice0<<<cdiv(NF, 256), 256, 0, stream>>>(x, HcatA, NF);

  // 5) layers
  float* hbuf[2] = {h0, h1};
  const float* cur = x;
  int b = 0;
  _Float16* Hc = HcatA;
  _Float16* Hn = HcatB;

  for (int l = 0; l <= NLAYERS; ++l) {
    const bool isOut = (l == NLAYERS);

    for (int k = 1; k <= KHOP; ++k) {
      float* nxt = hbuf[b];
      spmm_csr<<<cdiv((long long)N * 32, 256), 256, 0, stream>>>(
          rowptr, colsrc, colw, cur, nxt, Hc + (size_t)k * FDIM, N);
      cur = nxt;
      b ^= 1;
    }

    if (!isOut) {
      float* yout = hbuf[b];
      gemm_tag<8><<<gemm_blocks, 256, lds_hidden, stream>>>(
          Hc, Wt16 + (size_t)l * FDIM * KCAT, b_hidden + (size_t)l * FDIM,
          yout, Hn, nrowtiles, FDIM, FDIM, /*elu=*/1);
      cur = yout;
      b ^= 1;
      _Float16* t = Hc; Hc = Hn; Hn = t;
    } else {
      gemm_tag<1><<<gemm_blocks, 256, lds_out, stream>>>(
          Hc, WoutT, b_out, (float*)d_out, (_Float16*)nullptr,
          nrowtiles, COUT, COUT, /*elu=*/0);
    }
  }
}
